// TokenCondensation_11355893531164
// MI455X (gfx1250) — matrix operand order repaired and checked by reading, not compiled, not run on hardware
//
#include <hip/hip_runtime.h>

typedef _Float16 v16h __attribute__((ext_vector_type(16)));
typedef _Float16 half8 __attribute__((ext_vector_type(8)));
typedef float    v8f  __attribute__((ext_vector_type(8)));

#define B_  8
#define N_  4096
#define H_  512
#define M_  512
#define NHEAD 8
#define HD  64

// ---------------------------------------------------------------------------
// WMMA helpers (layouts per CDNA5 ISA §7.12.2, wave32)
// ---------------------------------------------------------------------------
__device__ __forceinline__ v8f wmma_f16(v16h a, v16h b, v8f c) {
    return __builtin_amdgcn_wmma_f32_16x16x32_f16(
        /*neg_a=*/false, a, /*neg_b=*/false, b,
        /*c_mod=*/(short)0, c, /*reuse_a=*/false, /*reuse_b=*/false);
}

// A-matrix 16x32 f16 fragment. tile points at element (row0, k0); element
// (r, k) lives at tile[r*ld + k].  Lane L holds row L%16; with base=8*(L>=16):
// h[0..7] = K base..base+7, h[8..15] = K 16+base..16+base+7.
__device__ __forceinline__ v16h load_afrag(const _Float16* tile, int ld, int lane) {
    const int r   = lane & 15;
    const int sel = (lane >> 4) & 1;
    const _Float16* p = tile + (size_t)r * ld + sel * 8;
    half8 lo = *(const half8*)(p);
    half8 hi = *(const half8*)(p + 16);
    v16h o;
#pragma unroll
    for (int i = 0; i < 8; ++i) { o[i] = lo[i]; o[i + 8] = hi[i]; }
    return o;
}

// B-matrix 32x16 f16 fragment. Element (k, col) lives at tile[col*ld + k].
// Lane L holds column L%16; with base=8*(L>=16): h[0..7] = K base..base+7, h[8..15] = K 16+base..16+base+7.
__device__ __forceinline__ v16h load_bfrag(const _Float16* tile, int ld, int lane) {
    const int c   = lane & 15;
    const int sel = (lane >> 4) & 1;
    const _Float16* p = tile + (size_t)c * ld + sel * 8;
    half8 lo = *(const half8*)(p);
    half8 hi = *(const half8*)(p + 16);
    v16h o;
#pragma unroll
    for (int i = 0; i < 8; ++i) { o[i] = lo[i]; o[i + 8] = hi[i]; }
    return o;
}

// ---------------------------------------------------------------------------
// DPP16 butterfly max-reduction across each 16-lane half (pure VALU, no LDS).
// quad_perm xor1 (0xB1), quad_perm xor2 (0x4E), ROW_HALF_MIRROR (0x141),
// ROW_MIRROR (0x140) -> every lane in a 16-group holds the group max.
// ---------------------------------------------------------------------------
template <int CTRL>
__device__ __forceinline__ float dppf(float x) {
    const int xi = __builtin_bit_cast(int, x);
    return __builtin_bit_cast(float,
        __builtin_amdgcn_update_dpp(xi, xi, CTRL, 0xF, 0xF, true));
}
__device__ __forceinline__ float redmax16(float x) {
    x = fmaxf(x, dppf<0xB1>(x));
    x = fmaxf(x, dppf<0x4E>(x));
    x = fmaxf(x, dppf<0x141>(x));
    x = fmaxf(x, dppf<0x140>(x));
    return x;
}

// ---------------------------------------------------------------------------
// 1) Row-normalize features (fp32 fn for FPS) + fp16 copy for WMMA GEMMs.
// ---------------------------------------------------------------------------
__global__ void normalize_kernel(const float* __restrict__ feat,
                                 float* __restrict__ fn,
                                 _Float16* __restrict__ feat_h) {
    const int row = blockIdx.x;
    const int tid = threadIdx.x;
    const float* src = feat + (size_t)row * H_;
    float a = src[tid];
    float b = src[tid + 256];
    __shared__ float red[256];
    red[tid] = a * a + b * b;
    __syncthreads();
    for (int s = 128; s > 0; s >>= 1) {
        if (tid < s) red[tid] += red[tid + s];
        __syncthreads();
    }
    const float inv = 1.0f / fmaxf(sqrtf(red[0]), 1e-12f);
    float* dstf = fn + (size_t)row * H_;
    _Float16* dsth = feat_h + (size_t)row * H_;
    dstf[tid]       = a * inv;
    dstf[tid + 256] = b * inv;
    dsth[tid]       = (_Float16)a;
    dsth[tid + 256] = (_Float16)b;
}

// ---------------------------------------------------------------------------
// 2) fp32 -> fp16 conversion of the four weight matrices.
// ---------------------------------------------------------------------------
__global__ void wconv_kernel(const float* __restrict__ wq, const float* __restrict__ wk,
                             const float* __restrict__ wv, const float* __restrict__ wo,
                             _Float16* __restrict__ wqh, _Float16* __restrict__ wkh,
                             _Float16* __restrict__ wvh, _Float16* __restrict__ woh) {
    const int i = blockIdx.x * blockDim.x + threadIdx.x;  // 0 .. 512*512-1
    wqh[i] = (_Float16)wq[i];
    wkh[i] = (_Float16)wk[i];
    wvh[i] = (_Float16)wv[i];
    woh[i] = (_Float16)wo[i];
}

// ---------------------------------------------------------------------------
// 3) Furthest point sampling. One block (512 threads) per batch.
//    min_d lives in registers (8 rows/thread); query vector staged in LDS.
//    Tie-break = lowest index (jnp.argmax first-occurrence semantics).
// ---------------------------------------------------------------------------
__global__ void fps_kernel(const float* __restrict__ fn,
                           const float* __restrict__ feat,
                           _Float16* __restrict__ cent_h,
                           int* __restrict__ idx_out) {
    const int b   = blockIdx.x;
    const int tid = threadIdx.x;  // 512
    __shared__ __align__(16) float q[H_];
    __shared__ float rv[512];
    __shared__ int   ri[512];

    const float* fnb = fn   + (size_t)b * N_ * H_;
    const float* ftb = feat + (size_t)b * N_ * H_;
    float mind[8];

    // t = 0: centroid 0 is token 0.
    if (tid == 0) idx_out[b * M_] = 0;
    cent_h[((size_t)b * M_) * H_ + tid] = (_Float16)ftb[tid];
    q[tid] = fnb[tid];
    __syncthreads();
#pragma unroll
    for (int k = 0; k < 8; ++k) {
        const int r = tid + (k << 9);
        const float4* rp = (const float4*)(fnb + (size_t)r * H_);
        const float4* qp = (const float4*)q;
        float acc = 0.0f;
#pragma unroll 4
        for (int i = 0; i < H_ / 4; ++i) {
            float4 x = rp[i], y = qp[i];
            acc += x.x * y.x + x.y * y.y + x.z * y.z + x.w * y.w;
        }
        mind[k] = 1.0f - acc;
    }
    __syncthreads();

    for (int t = 1; t < M_; ++t) {
        // block-wide argmax of min_d with lowest-index tie-break
        float bv = mind[0];
        int   bi = tid;
#pragma unroll
        for (int k = 1; k < 8; ++k) {
            const int r = tid + (k << 9);
            if (mind[k] > bv) { bv = mind[k]; bi = r; }
        }
        rv[tid] = bv;
        ri[tid] = bi;
        __syncthreads();
        for (int s = 256; s > 0; s >>= 1) {
            if (tid < s) {
                const float v2 = rv[tid + s];
                const int   i2 = ri[tid + s];
                if (v2 > rv[tid] || (v2 == rv[tid] && i2 < ri[tid])) {
                    rv[tid] = v2; ri[tid] = i2;
                }
            }
            __syncthreads();
        }
        const int idx = ri[0];
        if (tid == 0) idx_out[b * M_ + t] = idx;
        cent_h[((size_t)b * M_ + t) * H_ + tid] = (_Float16)ftb[(size_t)idx * H_ + tid];
        q[tid] = fnb[(size_t)idx * H_ + tid];
        __syncthreads();
#pragma unroll
        for (int k = 0; k < 8; ++k) {
            const int r = tid + (k << 9);
            const float4* rp = (const float4*)(fnb + (size_t)r * H_);
            const float4* qp = (const float4*)q;
            float acc = 0.0f;
#pragma unroll 4
            for (int i = 0; i < H_ / 4; ++i) {
                float4 x = rp[i], y = qp[i];
                acc += x.x * y.x + x.y * y.y + x.z * y.z + x.w * y.w;
            }
            mind[k] = fminf(mind[k], 1.0f - acc);
        }
        __syncthreads();
    }
}

// ---------------------------------------------------------------------------
// 4) WMMA GEMM:  C[r, j] = sum_k A[r,k] * W[j,k]   (torch Linear: x @ W.T)
//    One wave computes a 16x64 tile (A fragment reused across 4 WMMAs/k-step).
//    MODE 0: f16 out [Mtot, 512]
//    MODE 1: f16 out transposed per batch: out[b, j, n], n = r % rowsPerBatch
//    MODE 2: f32 out [Mtot, 512]
// ---------------------------------------------------------------------------
template <int MODE>
__global__ void gemm_aw_kernel(const _Float16* __restrict__ A,
                               const _Float16* __restrict__ W,
                               void* __restrict__ out,
                               int rowsPerBatch) {
    const int wid  = blockIdx.x * 8 + (threadIdx.x >> 5);
    const int lane = threadIdx.x & 31;
    const int mt = wid >> 3;   // row tile (16 rows)
    const int ng = wid & 7;    // 8 column groups of 64

    const _Float16* Abase = A + (size_t)(mt * 16) * H_;
    const _Float16* Wbase = W + (size_t)(ng * 64) * H_;

    v8f c0 = {}, c1 = {}, c2 = {}, c3 = {};
#pragma unroll 2
    for (int kk = 0; kk < H_; kk += 32) {
        if (kk + 32 < H_) {
            __builtin_prefetch(Abase + kk + 32, 0, 3);  // global_prefetch_b8
        }
        const v16h a = load_afrag(Abase + kk, H_, lane);
        c0 = wmma_f16(a, load_bfrag(Wbase + kk,                    H_, lane), c0);
        c1 = wmma_f16(a, load_bfrag(Wbase + (size_t)16 * H_ + kk, H_, lane), c1);
        c2 = wmma_f16(a, load_bfrag(Wbase + (size_t)32 * H_ + kk, H_, lane), c2);
        c3 = wmma_f16(a, load_bfrag(Wbase + (size_t)48 * H_ + kk, H_, lane), c3);
    }

    const int col = lane & 15;
    const int sel = lane >> 4;
#pragma unroll
    for (int v = 0; v < 8; ++v) {
        const int row = mt * 16 + v + sel * 8;
        float vals[4] = {c0[v], c1[v], c2[v], c3[v]};
#pragma unroll
        for (int g = 0; g < 4; ++g) {
            const int j = ng * 64 + g * 16 + col;
            if (MODE == 0) {
                ((_Float16*)out)[(size_t)row * H_ + j] = (_Float16)vals[g];
            } else if (MODE == 1) {
                const int bb = row / rowsPerBatch;
                const int n  = row % rowsPerBatch;
                ((_Float16*)out)[((size_t)bb * H_ + j) * rowsPerBatch + n] = (_Float16)vals[g];
            } else {
                ((float*)out)[(size_t)row * H_ + j] = vals[g];
            }
        }
    }
}

// ---------------------------------------------------------------------------
// 5) Flash-style cross attention. One wave per (batch, head, 16-row Q tile).
//    Streams keys in chunks of 32; online softmax with DPP16 max-reduction;
//    row sums computed by an extra WMMA against a ones matrix; P bounced
//    through per-wave LDS to convert C-layout -> A-layout for P·V.
//    Qh: [B, M, H] f16;  Kh: [B, N, H] f16;  Vt: [B, H, N] f16 (transposed).
// ---------------------------------------------------------------------------
__global__ void attn_kernel(const _Float16* __restrict__ Qh,
                            const _Float16* __restrict__ Kh,
                            const _Float16* __restrict__ Vt,
                            _Float16* __restrict__ aggh) {
    __shared__ __align__(16) _Float16 sP[8][16 * 32];  // per-wave 16x32 P tile
    const int wslot = threadIdx.x >> 5;
    const int lane  = threadIdx.x & 31;
    const int wid = blockIdx.x * 8 + wslot;
    const int b  = wid >> 8;          // 8*32 waves per batch
    const int h  = (wid >> 5) & 7;
    const int mt = wid & 31;

    const _Float16* Qtile = Qh + ((size_t)b * M_ + mt * 16) * H_ + h * HD;
    const v16h aq0 = load_afrag(Qtile,      H_, lane);   // d = 0..31
    const v16h aq1 = load_afrag(Qtile + 32, H_, lane);   // d = 32..63

    v16h ones;
#pragma unroll
    for (int i = 0; i < 16; ++i) ones[i] = (_Float16)1.0f;

    v8f acc0 = {}, acc1 = {}, acc2 = {}, acc3 = {};
    float mrow[8], lrow[8], alpha[8];
#pragma unroll
    for (int v = 0; v < 8; ++v) { mrow[v] = -__builtin_inff(); lrow[v] = 0.0f; }

    const float scale = 0.125f;  // 1/sqrt(64)
    const int   col = lane & 15;
    const int   sel = lane >> 4;
    const size_t bKV = (size_t)b * N_ * H_;
    const _Float16* Vhead = Vt + (size_t)b * H_ * N_ + (size_t)(h * HD) * N_;

    for (int n0 = 0; n0 < N_; n0 += 32) {
        const _Float16* Kt0 = Kh + bKV + (size_t)n0 * H_ + h * HD;
        const _Float16* Kt1 = Kt0 + (size_t)16 * H_;

        // prefetch next chunk's K rows and V segment (uniform branch)
        if (n0 + 32 < N_) {
            __builtin_prefetch(Kt0 + (size_t)32 * H_, 0, 3);
            __builtin_prefetch(Vhead + n0 + 32, 0, 3);
        }

        // --- S = Q · K^T  (16 rows x 32 keys), f32 accumulate ---
        v8f s0 = {}, s1 = {};
        s0 = wmma_f16(aq0, load_bfrag(Kt0,      H_, lane), s0);
        s0 = wmma_f16(aq1, load_bfrag(Kt0 + 32, H_, lane), s0);
        s1 = wmma_f16(aq0, load_bfrag(Kt1,      H_, lane), s1);
        s1 = wmma_f16(aq1, load_bfrag(Kt1 + 32, H_, lane), s1);

        // --- online max update (DPP16 reduction, no LDS), rescale, stage P ---
#pragma unroll
        for (int v = 0; v < 8; ++v) {
            float a0 = s0[v] * scale;
            float a1 = s1[v] * scale;
            const float t    = redmax16(fmaxf(a0, a1));
            const float mnew = fmaxf(mrow[v], t);
            alpha[v] = __expf(mrow[v] - mnew);
            mrow[v]  = mnew;
            a0 = __expf(a0 - mnew);
            a1 = __expf(a1 - mnew);
            acc0[v] *= alpha[v]; acc1[v] *= alpha[v];
            acc2[v] *= alpha[v]; acc3[v] *= alpha[v];
            const int prow = v + sel * 8;
            sP[wslot][prow * 32 + col]      = (_Float16)a0;
            sP[wslot][prow * 32 + 16 + col] = (_Float16)a1;
        }
        // same-wave LDS ordering; keep compiler from reordering across it
        asm volatile("s_wait_dscnt 0x0" ::: "memory");

        // --- P in A-layout; row sums via WMMA against ones matrix ---
        const v16h ap = load_afrag(&sP[wslot][0], 32, lane);
        v8f z = {};
        const v8f rsum = wmma_f16(ap, ones, z);  // every col = row sum of P
#pragma unroll
        for (int v = 0; v < 8; ++v) lrow[v] = lrow[v] * alpha[v] + rsum[v];

        // --- P · V chunk ---
        const _Float16* Vb = Vhead + n0;
        acc0 = wmma_f16(ap, load_bfrag(Vb + (size_t)0  * N_, N_, lane), acc0);
        acc1 = wmma_f16(ap, load_bfrag(Vb + (size_t)16 * N_, N_, lane), acc1);
        acc2 = wmma_f16(ap, load_bfrag(Vb + (size_t)32 * N_, N_, lane), acc2);
        acc3 = wmma_f16(ap, load_bfrag(Vb + (size_t)48 * N_, N_, lane), acc3);
    }

    // --- epilogue: divide by row sums, store f16 agg [B, M, H] ---
#pragma unroll
    for (int v = 0; v < 8; ++v) {
        const float inv = 1.0f / lrow[v];
        const int row = mt * 16 + v + sel * 8;
        const size_t base = ((size_t)b * M_ + row) * H_ + h * HD + col;
        aggh[base + 0]  = (_Float16)(acc0[v] * inv);
        aggh[base + 16] = (_Float16)(acc1[v] * inv);
        aggh[base + 32] = (_Float16)(acc2[v] * inv);
        aggh[base + 48] = (_Float16)(acc3[v] * inv);
    }
}

// ---------------------------------------------------------------------------
// Launch
// ---------------------------------------------------------------------------
extern "C" void kernel_launch(void* const* d_in, const int* in_sizes, int n_in,
                              void* d_out, int out_size, void* d_ws, size_t ws_size,
                              hipStream_t stream) {
    (void)in_sizes; (void)n_in; (void)out_size; (void)ws_size;

    const float* feat = (const float*)d_in[0];
    const float* Wq   = (const float*)d_in[1];
    const float* Wk   = (const float*)d_in[2];
    const float* Wv   = (const float*)d_in[3];
    const float* Wo   = (const float*)d_in[4];

    float* out_f   = (float*)d_out;                        // [B, M, H] f32
    int*   idx_out = (int*)(out_f + (size_t)B_ * M_ * H_); // [B, M] int32

    // workspace carve-up (all chunks 256B-aligned by construction)
    char* ws = (char*)d_ws;
    size_t off = 0;
    float*    fn     = (float*)   (ws + off); off += (size_t)B_ * N_ * H_ * 4;  // 64 MB
    _Float16* feat_h = (_Float16*)(ws + off); off += (size_t)B_ * N_ * H_ * 2;  // 32 MB
    _Float16* Wq_h   = (_Float16*)(ws + off); off += (size_t)H_ * H_ * 2;
    _Float16* Wk_h   = (_Float16*)(ws + off); off += (size_t)H_ * H_ * 2;
    _Float16* Wv_h   = (_Float16*)(ws + off); off += (size_t)H_ * H_ * 2;
    _Float16* Wo_h   = (_Float16*)(ws + off); off += (size_t)H_ * H_ * 2;
    _Float16* cent_h = (_Float16*)(ws + off); off += (size_t)B_ * M_ * H_ * 2;  // 4 MB
    _Float16* Q_h    = (_Float16*)(ws + off); off += (size_t)B_ * M_ * H_ * 2;  // 4 MB
    _Float16* K_h    = (_Float16*)(ws + off); off += (size_t)B_ * N_ * H_ * 2;  // 32 MB
    _Float16* V_t    = (_Float16*)(ws + off); off += (size_t)B_ * N_ * H_ * 2;  // 32 MB (transposed)
    _Float16* agg_h  = (_Float16*)(ws + off); off += (size_t)B_ * M_ * H_ * 2;  // 4 MB

    // 1) normalize + f16 copy of features
    normalize_kernel<<<B_ * N_, 256, 0, stream>>>(feat, fn, feat_h);

    // 2) f16 weights
    wconv_kernel<<<(H_ * H_) / 256, 256, 0, stream>>>(Wq, Wk, Wv, Wo,
                                                      Wq_h, Wk_h, Wv_h, Wo_h);

    // 3) FPS (writes centroid gather + indices)
    fps_kernel<<<B_, 512, 0, stream>>>(fn, feat, cent_h, idx_out);

    // 4) projections (WMMA, 16x64 tile/wave): Q = cent@Wq^T, K = feat@Wk^T, V^T
    gemm_aw_kernel<0><<<(B_ * M_) / 16, 256, 0, stream>>>(cent_h, Wq_h, Q_h, M_);
    gemm_aw_kernel<0><<<(B_ * N_) / 16, 256, 0, stream>>>(feat_h, Wk_h, K_h, N_);
    gemm_aw_kernel<1><<<(B_ * N_) / 16, 256, 0, stream>>>(feat_h, Wv_h, V_t, N_);

    // 5) flash cross-attention (WMMA): 8*8*32 = 2048 waves, 8 waves/block
    attn_kernel<<<(B_ * NHEAD * (M_ / 16)) / 8, 256, 0, stream>>>(Q_h, K_h, V_t, agg_h);

    // 6) output projection (WMMA, f32 out): out = agg @ Wo^T
    gemm_aw_kernel<2><<<(B_ * M_) / 16, 256, 0, stream>>>(agg_h, Wo_h, out_f, M_);
}
